// Attention_45638322487965
// MI455X (gfx1250) — matrix operand order repaired and checked by reading, not compiled, not run on hardware
//
#include <hip/hip_runtime.h>
#include <hip/hip_bf16.h>

// ---------------------------------------------------------------------------
// CDNA5 (gfx1250) attention: bf16 WMMA GEMMs + flash attention, wave32.
// Fragment reads: contiguous per lane -> ds_load_b128 pairs.
// Flash Q/K staging: global_load_async_to_lds_b128 (ASYNCcnt path).
// bf16 conversion via native fptrunc (hardware cvt ops, RTNE).
// ---------------------------------------------------------------------------

typedef __attribute__((ext_vector_type(16))) __bf16 v16bf;
typedef __attribute__((ext_vector_type(8)))  float  v8f;

#define K_MASK (-2.3819763e38f)

// ----- bf16 helpers (hardware conversion via fptrunc) -----------------------
__device__ __forceinline__ unsigned short f2bfbits(float f) {
  return __builtin_bit_cast(unsigned short, (__bf16)f);
}
__device__ __forceinline__ float bfbits2f(unsigned short h) {
  unsigned u = ((unsigned)h) << 16;
  return __builtin_bit_cast(float, u);
}
__device__ __forceinline__ __bf16 bits2bf(unsigned short h) {
  return __builtin_bit_cast(__bf16, h);
}
__device__ __forceinline__ unsigned pack2(float lo, float hi) {
  return (unsigned)f2bfbits(lo) | ((unsigned)f2bfbits(hi) << 16);
}
// LDS generic pointers carry the LDS byte offset in the low 32 bits (ISA 10.2)
__device__ __forceinline__ unsigned lds_off(const void* p) {
  return (unsigned)(unsigned long long)(uintptr_t)p;
}
// Async copy of 16 bytes global -> LDS (GLOBAL_LOAD_ASYNC_TO_LDS_B128)
__device__ __forceinline__ void async_copy_b128(const void* gsrc, unsigned ldst) {
  asm volatile("global_load_async_to_lds_b128 %0, %1, off"
               :: "v"(ldst), "v"(gsrc) : "memory");
}
__device__ __forceinline__ void wait_async0() {
  asm volatile("s_wait_asynccnt 0" ::: "memory");
}

// ----- WMMA fragment loaders (16-bit, contiguous k order in A and B, wave32) -----------
// A (16x32, MxK): row m = lane%16, half = lane/16; one contiguous 16-elem run.
__device__ __forceinline__ v16bf load_a_frag(const unsigned short* sm,
                                             int mbase, int kbase, int ld) {
  int lane = threadIdx.x & 31;
  int m = mbase + (lane & 15);
  int half = lane >> 4;
  v16bf a;
#pragma unroll
  for (int i = 0; i < 16; ++i) {
    int g = i >> 1, p = i & 1;
    int k = 16 * half + 2 * g + p;
    a[i] = bits2bf(sm[m * ld + kbase + k]);
  }
  return a;
}
// B (KxN) staged n-major: element B[k][n] at sm[n*ld + k]; one 32B run/lane.
__device__ __forceinline__ v16bf load_b_frag_nmajor(const unsigned short* sm,
                                                    int kbase, int nbase, int ld) {
  int lane = threadIdx.x & 31;
  int n = nbase + (lane & 15);
  int half = lane >> 4;
  v16bf b;
#pragma unroll
  for (int i = 0; i < 16; ++i) {
    int g = i >> 1, p = i & 1;
    int k = 16 * half + 2 * g + p;
    b[i] = bits2bf(sm[n * ld + kbase + k]);
  }
  return b;
}

// ---------------------------------------------------------------------------
// GEMM: C[M,N] = A[M,K] (fp32 or bf16) * B[K,N] (fp32 -> bf16).
// Block: 128 threads = 4 waves. Block tile 64x64, wave tile 16x64, K step 32.
// ---------------------------------------------------------------------------
__global__ __launch_bounds__(128)
void gemm_bf16_wmma(const float* __restrict__ Af32,
                    const unsigned short* __restrict__ Abf,
                    const float* __restrict__ B,
                    unsigned short* __restrict__ Cbf,
                    float* __restrict__ Cf32,
                    int M, int N, int K) {
  __shared__ __align__(16) unsigned short As[64 * 32];   // [m][k]
  __shared__ __align__(16) unsigned short Bst[64 * 32];  // [n][k]
  const int tid = threadIdx.x;
  const int wid = tid >> 5;
  const int m0 = blockIdx.y * 64;
  const int n0 = blockIdx.x * 64;

  // Static staging decomposition (no per-iteration divides / exec masking)
  const int ar = tid >> 2;          // A row        0..31 (+32 second pass)
  const int ac = (tid & 3) * 8;     // A col chunk  0,8,16,24
  const int br = tid >> 3;          // B k-row      0..15 (+16 second pass)
  const int bc = (tid & 7) * 8;     // B n-chunk    0..56

  v8f acc[4];
#pragma unroll
  for (int i = 0; i < 4; ++i)
#pragma unroll
    for (int j = 0; j < 8; ++j) acc[i][j] = 0.0f;

  for (int k0 = 0; k0 < K; k0 += 32) {
    __syncthreads();
    if (Af32) {
#pragma unroll
      for (int pass = 0; pass < 2; ++pass) {
        int r = ar + pass * 32;
        const float* src = Af32 + (size_t)(m0 + r) * K + k0 + ac;
        float4 f0 = *(const float4*)src;
        float4 f1 = *(const float4*)(src + 4);
        uint4 o;
        o.x = pack2(f0.x, f0.y); o.y = pack2(f0.z, f0.w);
        o.z = pack2(f1.x, f1.y); o.w = pack2(f1.z, f1.w);
        *(uint4*)&As[r * 32 + ac] = o;
      }
    } else {
#pragma unroll
      for (int pass = 0; pass < 2; ++pass) {
        int r = ar + pass * 32;
        *(uint4*)&As[r * 32 + ac] =
            *(const uint4*)(Abf + (size_t)(m0 + r) * K + k0 + ac);
      }
    }
#pragma unroll
    for (int pass = 0; pass < 2; ++pass) {
      int r = br + pass * 16;  // k row
      const float* src = B + (size_t)(k0 + r) * N + n0 + bc;
      float4 f0 = *(const float4*)src;
      float4 f1 = *(const float4*)(src + 4);
      float f[8] = {f0.x, f0.y, f0.z, f0.w, f1.x, f1.y, f1.z, f1.w};
#pragma unroll
      for (int j = 0; j < 8; ++j) Bst[(bc + j) * 32 + r] = f2bfbits(f[j]);
    }
    __syncthreads();

    v16bf a = load_a_frag(As, wid * 16, 0, 32);
#pragma unroll
    for (int nt = 0; nt < 4; ++nt) {
      v16bf bb = load_b_frag_nmajor(Bst, 0, nt * 16, 32);
      acc[nt] = __builtin_amdgcn_wmma_f32_16x16x32_bf16(
          false, a, false, bb, (short)0, acc[nt], false, false);
    }
  }

  const int lane = tid & 31;
  const int nloc = lane & 15;
  const int half = lane >> 4;
#pragma unroll
  for (int nt = 0; nt < 4; ++nt) {
#pragma unroll
    for (int r = 0; r < 8; ++r) {
      int m = m0 + wid * 16 + r + 8 * half;
      int n = n0 + nt * 16 + nloc;
      float v = acc[nt][r];
      if (Cbf) Cbf[(size_t)m * N + n] = f2bfbits(v);
      else     Cf32[(size_t)m * N + n] = v;
    }
  }
}

// ---------------------------------------------------------------------------
// RoPE (in-place on bf16 [B,T,nheads,128]), theta = 500000
// ---------------------------------------------------------------------------
__global__ void rope_kernel(unsigned short* __restrict__ qk,
                            const int* __restrict__ segpos,
                            int Bdim, int T, int nheads) {
  long long idx = (long long)blockIdx.x * blockDim.x + threadIdx.x;
  long long total = (long long)Bdim * T * nheads * 64;
  if (idx >= total) return;
  int i = (int)(idx & 63);
  long long tmp = idx >> 6;
  int h = (int)(tmp % nheads); tmp /= nheads;
  int t = (int)(tmp % T);
  int b = (int)(tmp / T);
  float pos = (float)segpos[b * T + t];
  float inv_freq = __expf(-(2.0f * (float)i / 128.0f) * __logf(500000.0f));
  float ang = pos * inv_freq;
  float s, c;
  __sincosf(ang, &s, &c);
  size_t base = ((size_t)(b * T + t) * nheads + h) * 128;
  float x1 = bfbits2f(qk[base + i]);
  float x2 = bfbits2f(qk[base + 64 + i]);
  qk[base + i]      = f2bfbits(x1 * c - x2 * s);
  qk[base + 64 + i] = f2bfbits(x2 * c + x1 * s);
}

// ---------------------------------------------------------------------------
// Flash attention (causal, GQA). Block: 128 threads = 4 waves, 64 q rows.
// Q/K staged via async-to-LDS DMA; V staged transposed (n-major) manually.
// ---------------------------------------------------------------------------
__global__ __launch_bounds__(128)
void flash_attn(const unsigned short* __restrict__ Q,
                const unsigned short* __restrict__ Kb,
                const unsigned short* __restrict__ Vb,
                unsigned short* __restrict__ Oout,
                int Bdim, int T, int NQ, int NKV) {
  const int H = 128;
  __shared__ __align__(16) unsigned short Qs[64 * 128];   // [t][h]  16 KB
  __shared__ __align__(16) unsigned short Ks[32 * 128];   // [s][h]   8 KB
  __shared__ __align__(16) unsigned short Vt[128 * 32];   // [h][s]   8 KB
  __shared__ __align__(16) unsigned short Ps[4][16 * 32]; // per-wave 4 KB

  const int q0   = blockIdx.x * 64;
  const int qh   = blockIdx.y;
  const int b    = blockIdx.z;
  const int kvh  = qh / (NQ / NKV);
  const int tid  = threadIdx.x;
  const int wid  = tid >> 5;
  const int lane = tid & 31;
  const int nloc = lane & 15;
  const int half = lane >> 4;
  const float scale = 0.08838834764831845f; // 1/sqrt(128)

  // Stage Q tile (64x128) via async DMA: 8 x 16-byte chunks per thread
  {
    const int qr = tid >> 4;          // 0..7 (+8 per pass)
    const int qc = (tid & 15) * 8;    // 0..120
#pragma unroll
    for (int pass = 0; pass < 8; ++pass) {
      int r = qr + pass * 8;
      const unsigned short* src =
          &Q[((size_t)(b * T + q0 + r) * NQ + qh) * H + qc];
      async_copy_b128(src, lds_off(&Qs[r * 128 + qc]));
    }
  }

  v8f O[8];
#pragma unroll
  for (int i = 0; i < 8; ++i)
#pragma unroll
    for (int j = 0; j < 8; ++j) O[i][j] = 0.0f;
  float mrun[8], lrun[8];
#pragma unroll
  for (int r = 0; r < 8; ++r) { mrun[r] = -3.0e38f; lrun[r] = 0.0f; }

  const int kr = tid >> 4;        // 0..7 (+8,16,24 per pass)
  const int kc = (tid & 15) * 8;  // 0..120

  const int s_end = q0 + 64;  // causal bound
  for (int s0 = 0; s0 < s_end; s0 += 32) {
    __syncthreads();  // previous tiles fully consumed
    // K tile (32x128) via async DMA
#pragma unroll
    for (int pass = 0; pass < 4; ++pass) {
      int s = kr + pass * 8;
      const unsigned short* src =
          &Kb[((size_t)(b * T + s0 + s) * NKV + kvh) * H + kc];
      async_copy_b128(src, lds_off(&Ks[s * 128 + kc]));
    }
    // V tile transposed -> Vt[h][s] (coalesced 128-bit reads, b16 stores)
#pragma unroll
    for (int pass = 0; pass < 4; ++pass) {
      int s = kr + pass * 8;
      uint4 vv = *(const uint4*)
          &Vb[((size_t)(b * T + s0 + s) * NKV + kvh) * H + kc];
      unsigned short hv[8] = {
          (unsigned short)(vv.x & 0xffff), (unsigned short)(vv.x >> 16),
          (unsigned short)(vv.y & 0xffff), (unsigned short)(vv.y >> 16),
          (unsigned short)(vv.z & 0xffff), (unsigned short)(vv.z >> 16),
          (unsigned short)(vv.w & 0xffff), (unsigned short)(vv.w >> 16)};
#pragma unroll
      for (int j = 0; j < 8; ++j) Vt[(kc + j) * 32 + s] = hv[j];
    }
    wait_async0();     // this wave's async writes are in LDS
    __syncthreads();   // all waves' tiles published

    // scores = Q * K^T  (16 x 32 per wave); B = K^T is n-major in Ks
    v8f sc0, sc1;
#pragma unroll
    for (int j = 0; j < 8; ++j) { sc0[j] = 0.0f; sc1[j] = 0.0f; }
#pragma unroll
    for (int kk = 0; kk < 4; ++kk) {
      v16bf a = load_a_frag(Qs, wid * 16, kk * 32, 128);
      v16bf b0 = load_b_frag_nmajor(Ks, kk * 32, 0, 128);
      sc0 = __builtin_amdgcn_wmma_f32_16x16x32_bf16(
          false, a, false, b0, (short)0, sc0, false, false);
      v16bf b1 = load_b_frag_nmajor(Ks, kk * 32, 16, 128);
      sc1 = __builtin_amdgcn_wmma_f32_16x16x32_bf16(
          false, a, false, b1, (short)0, sc1, false, false);
    }

    // Online softmax per accumulator row (global row = q0+wid*16+r+8*half)
#pragma unroll
    for (int r = 0; r < 8; ++r) {
      int row = q0 + wid * 16 + r + 8 * half;
      float v0 = sc0[r] * scale;
      float v1 = sc1[r] * scale;
      if (s0 + nloc      > row) v0 = K_MASK;
      if (s0 + 16 + nloc > row) v1 = K_MASK;
      float tmax = fmaxf(v0, v1);
#pragma unroll
      for (int d = 8; d >= 1; d >>= 1)
        tmax = fmaxf(tmax, __shfl_xor(tmax, d, 32));
      float mnew = fmaxf(mrun[r], tmax);
      float corr = __expf(mrun[r] - mnew);
      float p0 = __expf(v0 - mnew);
      float p1 = __expf(v1 - mnew);
      float psum = p0 + p1;
#pragma unroll
      for (int d = 8; d >= 1; d >>= 1) psum += __shfl_xor(psum, d, 32);
      lrun[r] = lrun[r] * corr + psum;
      mrun[r] = mnew;
#pragma unroll
      for (int ht = 0; ht < 8; ++ht) O[ht][r] *= corr;
      // C/D layout -> per-wave LDS scratch (wave-local DS ops are in-order)
      int m = r + 8 * half;
      Ps[wid][m * 32 + nloc]      = f2bfbits(p0);
      Ps[wid][m * 32 + 16 + nloc] = f2bfbits(p1);
    }

    // O += P * V   (16x32 @ 32x128); B = V with k=s: n-major in Vt
    v16bf pa = load_a_frag(&Ps[wid][0], 0, 0, 32);
#pragma unroll
    for (int ht = 0; ht < 8; ++ht) {
      v16bf vb = load_b_frag_nmajor(Vt, 0, ht * 16, 32);
      O[ht] = __builtin_amdgcn_wmma_f32_16x16x32_bf16(
          false, pa, false, vb, (short)0, O[ht], false, false);
    }
  }

  // Epilogue: divide by l and store bf16 [B,T,NQ,H]
#pragma unroll
  for (int r = 0; r < 8; ++r) {
    float inv = 1.0f / lrun[r];
    int t = q0 + wid * 16 + r + 8 * half;
#pragma unroll
    for (int ht = 0; ht < 8; ++ht) {
      int hcol = ht * 16 + nloc;
      Oout[((size_t)(b * T + t) * NQ + qh) * H + hcol] = f2bfbits(O[ht][r] * inv);
    }
  }
}

// ---------------------------------------------------------------------------
// Launch
// ---------------------------------------------------------------------------
extern "C" void kernel_launch(void* const* d_in, const int* in_sizes, int n_in,
                              void* d_out, int out_size, void* d_ws, size_t ws_size,
                              hipStream_t stream) {
  (void)in_sizes; (void)n_in; (void)out_size; (void)ws_size;
  const int B = 2, T = 2048, D = 4096, NQ = 32, NKV = 8, H = 128;
  const int M = B * T;     // 4096
  const int Nq = NQ * H;   // 4096
  const int Nkv = NKV * H; // 1024

  const float* x   = (const float*)d_in[0];
  const int*   seg = (const int*)d_in[1];
  // d_in[2] = attn_mask (causal; handled analytically)
  const float* wq  = (const float*)d_in[3];
  const float* wk  = (const float*)d_in[4];
  const float* wv  = (const float*)d_in[5];
  const float* wo  = (const float*)d_in[6];
  float* out = (float*)d_out;

  unsigned short* qbf = (unsigned short*)d_ws;          // 32 MB
  unsigned short* kbf = qbf + (size_t)M * Nq;           //  8 MB
  unsigned short* vbf = kbf + (size_t)M * Nkv;          //  8 MB
  unsigned short* qkv = vbf + (size_t)M * Nkv;          // 32 MB

  // 1) Projections (fp32 in -> bf16 out), WMMA bf16
  gemm_bf16_wmma<<<dim3(Nq / 64,  M / 64), 128, 0, stream>>>(
      x, nullptr, wq, qbf, nullptr, M, Nq, D);
  gemm_bf16_wmma<<<dim3(Nkv / 64, M / 64), 128, 0, stream>>>(
      x, nullptr, wk, kbf, nullptr, M, Nkv, D);
  gemm_bf16_wmma<<<dim3(Nkv / 64, M / 64), 128, 0, stream>>>(
      x, nullptr, wv, vbf, nullptr, M, Nkv, D);

  // 2) RoPE on q and k
  {
    long long nq = (long long)B * T * NQ * 64;
    long long nk = (long long)B * T * NKV * 64;
    rope_kernel<<<dim3((unsigned)((nq + 255) / 256)), 256, 0, stream>>>(
        qbf, seg, B, T, NQ);
    rope_kernel<<<dim3((unsigned)((nk + 255) / 256)), 256, 0, stream>>>(
        kbf, seg, B, T, NKV);
  }

  // 3) Flash attention (causal, GQA 4:1)
  flash_attn<<<dim3(T / 64, NQ, B), 128, 0, stream>>>(
      qbf, kbf, vbf, qkv, B, T, NQ, NKV);

  // 4) Output projection (bf16 A from ws, fp32 wo, fp32 out)
  gemm_bf16_wmma<<<dim3(D / 64, M / 64), 128, 0, stream>>>(
      nullptr, qkv, wo, nullptr, out, M, D, Nq);
}
